// AAGF_69939247448724
// MI455X (gfx1250) — matrix operand order repaired and checked by reading, not compiled, not run on hardware
//
#include <hip/hip_runtime.h>
#include <stdint.h>

// ---------------------------------------------------------------------------
// Types for CDNA5 WMMA (wave32): v_wmma_f32_16x16x32_bf16
// ---------------------------------------------------------------------------
typedef __bf16 bf16;
typedef __attribute__((ext_vector_type(16))) __bf16 bfv16;
typedef __attribute__((ext_vector_type(8)))  __bf16 bfv8;
typedef __attribute__((ext_vector_type(8)))  float  fv8;
typedef __attribute__((__vector_size__(4 * sizeof(int)))) int v4i;

#define BN_SCALE 0.9999950000374997f  // 1/sqrt(1+1e-5)

#ifndef __has_builtin
#define __has_builtin(x) 0
#endif

#if __has_builtin(__builtin_amdgcn_global_load_async_to_lds_b128)
#define HAVE_ASYNC 1
#else
#define HAVE_ASYNC 0
#endif

#if __has_builtin(__builtin_amdgcn_s_wait_asynccnt)
#define HAVE_WAIT_ASYNC_BUILTIN 1
#else
#define HAVE_WAIT_ASYNC_BUILTIN 0
#endif

#define AS1 __attribute__((address_space(1)))
#define AS3 __attribute__((address_space(3)))

__device__ __forceinline__ bf16 f2bf(float f) {
  uint32_t u = __builtin_bit_cast(uint32_t, f);
  u += 0x7FFFu + ((u >> 16) & 1u);           // round-to-nearest-even
  uint16_t h = (uint16_t)(u >> 16);
  return __builtin_bit_cast(bf16, h);
}

// 16B global -> LDS copy: async DMA path on CDNA5, sync fallback otherwise.
__device__ __forceinline__ void cp16(void* l, const void* g) {
#if HAVE_ASYNC
  __builtin_amdgcn_global_load_async_to_lds_b128(
      (AS1 v4i*)(g), (AS3 v4i*)(l), 0, 0);
#else
  *(bfv8*)l = *(const bfv8*)g;
#endif
}

__device__ __forceinline__ void wait_async() {
#if HAVE_WAIT_ASYNC_BUILTIN
  __builtin_amdgcn_s_wait_asynccnt(0);
#elif HAVE_ASYNC
  asm volatile("s_wait_asynccnt 0x0" ::: "memory");
#endif
}

// ---------------------------------------------------------------------------
// Generic implicit-GEMM conv (KHxKW, stride 1), bf16 in, f32 accumulate.
// Workgroup: 256 threads = 8 waves; tile = 64 pixels x 64 Cout, K staged
// 64-deep. Wave (wm 0..3, wn 0..1) computes 16 px x 32 cout via a 2(K)x2(N)
// unrolled WMMA block per staging round (4 v_wmma per barrier pair; A frag
// reused across both N subtiles).
// Epilogue: +bias, optional BN (g*BN_SCALE, b), optional ReLU, optional
// residual add; writes f32 and/or bf16.
// ---------------------------------------------------------------------------
__global__ __launch_bounds__(256, 2)
void conv_wmma(const bf16* __restrict__ x, int nImg, int Himg, int Wimg, int Cin,
               const bf16* __restrict__ wT,        // [taps][Cout][Cin] bf16
               const float* __restrict__ bias,
               const float* __restrict__ bnG, const float* __restrict__ bnB,
               int Cout, int KH, int KW, int pad, int relu,
               float* __restrict__ outF, bf16* __restrict__ outH,
               const float* __restrict__ residual)
{
  __shared__ __align__(16) bf16 At[64][64];   // [pixel][ci]
  __shared__ __align__(16) bf16 Bt[64][64];   // [co][ci] (transposed weight tile)

  const int tid = threadIdx.x;
  const int HW = Himg * Wimg;
  const int tilesPerImg = (HW + 63) >> 6;
  const int img     = blockIdx.x / tilesPerImg;
  const int pixBase = (blockIdx.x % tilesPerImg) << 6;
  const int coBase  = blockIdx.y << 6;

  const int wave   = tid >> 5;
  const int lane   = tid & 31;
  const int wm     = wave & 3;      // M subtile (16 pixels each)
  const int wn     = wave >> 2;     // N half (32 couts each)
  const int laneHi = lane >> 4;     // 0 or 1
  const int lane16 = lane & 15;

  // staging decode: thread t -> row sr (0..63), 16-elem quad sq (0..3)
  const int sr = tid >> 2, sq = tid & 3;

  fv8 acc0 = {}, acc1 = {};

  const int taps = KH * KW;
  for (int tap = 0; tap < taps; ++tap) {
    const int ky = tap / KW, kx = tap % KW;
    for (int c0 = 0; c0 < Cin; c0 += 64) {
      __syncthreads();   // protect LDS WAR vs previous round's compute
      // ---- stage A tile (64 px x 64 ci), zero-padded at borders ----
      {
        const int pix = pixBase + sr;
        bool valid = false;
        const bf16* src = nullptr;
        if (pix < HW) {
          const int py = pix / Wimg, px = pix - py * Wimg;
          const int sy = py + ky - pad, sx = px + kx - pad;
          if ((unsigned)sy < (unsigned)Himg && (unsigned)sx < (unsigned)Wimg) {
            valid = true;
            src = x + (((size_t)img * HW + (size_t)sy * Wimg + sx) * Cin
                       + c0 + sq * 16);
          }
        }
        if (valid) {
          cp16(&At[sr][sq * 16],     src);
          cp16(&At[sr][sq * 16 + 8], src + 8);
        } else {
          bfv8 z = {};
          *(bfv8*)(&At[sr][sq * 16])     = z;
          *(bfv8*)(&At[sr][sq * 16 + 8]) = z;
        }
      }
      // ---- stage B tile (64 co x 64 ci), from pre-transposed weights ----
      {
        const bf16* src = wT + (((size_t)tap * Cout + coBase + sr) * Cin
                                + c0 + sq * 16);
        cp16(&Bt[sr][sq * 16],     src);
        cp16(&Bt[sr][sq * 16 + 8], src + 8);
      }
      wait_async();
      __syncthreads();
      // ---- 2(K) x 2(N) WMMA block; ISA 16-bit A(16x32)/B(32x16) layouts ----
      {
        const int m  = (wm << 4) + lane16;
        const int nb = wn << 5;
#pragma unroll
        for (int kc = 0; kc < 64; kc += 32) {
          bfv8 a0 = *(const bfv8*)(&At[m][kc + laneHi * 8]);       // K 0..7 / 8..15
          bfv8 a1 = *(const bfv8*)(&At[m][kc + 16 + laneHi * 8]);  // K 16..23 / 24..31
          bfv16 av;
#pragma unroll
          for (int i = 0; i < 8; ++i) { av[i] = a0[i]; av[i + 8] = a1[i]; }
#pragma unroll
          for (int j = 0; j < 2; ++j) {
            const int n = nb + (j << 4) + lane16;
            bfv8 b0 = *(const bfv8*)(&Bt[n][kc + laneHi * 8]);       // K 0..7 / 8..15
            bfv8 b1 = *(const bfv8*)(&Bt[n][kc + 16 + laneHi * 8]);  // K 16..23 / 24..31
            bfv16 bv;
#pragma unroll
            for (int i = 0; i < 8; ++i) { bv[i] = b0[i]; bv[i + 8] = b1[i]; }
            if (j == 0) {
              acc0 = __builtin_amdgcn_wmma_f32_16x16x32_bf16(
                  false, av, false, bv, (short)0, acc0, false, false);
            } else {
              acc1 = __builtin_amdgcn_wmma_f32_16x16x32_bf16(
                  false, av, false, bv, (short)0, acc1, false, false);
            }
          }
        }
      }
    }
  }

  // ---- epilogue (two 16-cout subtiles per wave) ----
#pragma unroll
  for (int j = 0; j < 2; ++j) {
    const int n = coBase + (wn << 5) + (j << 4) + lane16;
    const float bia = bias ? bias[n] : 0.f;
    const float gg  = bnG ? bnG[n] * BN_SCALE : 1.f;
    const float sh  = bnB ? bnB[n] : 0.f;
    const fv8 acc = j ? acc1 : acc0;
#pragma unroll
    for (int r = 0; r < 8; ++r) {
      const int mloc = (wm << 4) + r + laneHi * 8;
      const int pix = pixBase + mloc;
      if (pix >= HW) continue;
      float v = acc[r] + bia;
      if (bnG) v = v * gg + sh;
      if (relu) v = fmaxf(v, 0.f);
      const size_t o = ((size_t)img * HW + pix) * Cout + n;
      if (residual) v += residual[o];
      if (outF) outF[o] = v;
      if (outH) outH[o] = f2bf(v);
    }
  }
}

// ---------------------------------------------------------------------------
// Weight prep: f32 [taps][Cin][Cout] -> bf16 [taps][Cout][Cin]
// ---------------------------------------------------------------------------
__global__ void wprep(const float* __restrict__ w, bf16* __restrict__ wT,
                      int Cin, int Cout, long total)
{
  long i = (long)blockIdx.x * 256 + threadIdx.x;
  if (i >= total) return;
  int co = (int)(i % Cout);
  long t1 = i / Cout;
  int ci = (int)(t1 % Cin);
  int tap = (int)(t1 / Cin);
  wT[((size_t)tap * Cout + co) * Cin + ci] = f2bf(w[i]);
}

__global__ void tobf(const float* __restrict__ s, bf16* __restrict__ d, long n)
{
  long i = (long)blockIdx.x * 256 + threadIdx.x;
  if (i < n) d[i] = f2bf(s[i]);
}

// write f32 src (pix-major, C chans) into bf16 dst with Ctot chans at Coff
__global__ void catw(const float* __restrict__ s, bf16* __restrict__ d,
                     int C, int Ctot, int Coff, long n)
{
  long i = (long)blockIdx.x * 256 + threadIdx.x;
  if (i >= n) return;
  long pix = i / C;
  int c = (int)(i % C);
  d[pix * Ctot + Coff + c] = f2bf(s[i]);
}

__global__ void avgk(const float* __restrict__ a, const float* __restrict__ b,
                     float* __restrict__ o, long n)
{
  long i = (long)blockIdx.x * 256 + threadIdx.x;
  if (i < n) o[i] = 0.5f * (a[i] + b[i]);
}

// detector head: amap = sigmoid(h . w + b), h is (B*HW, 64)
__global__ void detheadk(const float* __restrict__ h, const float* __restrict__ w,
                         const float* __restrict__ b, float* __restrict__ amap)
{
  int pix = blockIdx.x * 256 + threadIdx.x;
  const float* hp = h + (size_t)pix * 64;
  float s = b[0];
#pragma unroll
  for (int c = 0; c < 64; ++c) s += hp[c] * w[c];
  amap[pix] = 1.f / (1.f + __expf(-s));
}

// per-batch top-16 (JAX semantics: sorted desc, lowest index on ties) + sum
__global__ void topkk(const float* __restrict__ amap, int HW, int Wimg,
                      int* __restrict__ coords, float* __restrict__ sums)
{
  __shared__ float vals[4096];
  __shared__ float red[256];
  __shared__ int   redi[256];
  const float* am = amap + (size_t)blockIdx.x * HW;
  const int tid = threadIdx.x;
  float s = 0.f;
  for (int i = tid; i < HW; i += 256) { float v = am[i]; vals[i] = v; s += v; }
  red[tid] = s; __syncthreads();
  for (int off = 128; off > 0; off >>= 1) {
    if (tid < off) red[tid] += red[tid + off];
    __syncthreads();
  }
  if (tid == 0) sums[blockIdx.x] = red[0];
  __syncthreads();
  for (int k = 0; k < 16; ++k) {
    float bv = -1e30f; int bi = HW;
    for (int i = tid; i < HW; i += 256) {
      float v = vals[i];
      if (v > bv || (v == bv && i < bi)) { bv = v; bi = i; }
    }
    red[tid] = bv; redi[tid] = bi; __syncthreads();
    for (int off = 128; off > 0; off >>= 1) {
      if (tid < off) {
        float ov = red[tid + off]; int oi = redi[tid + off];
        if (ov > red[tid] || (ov == red[tid] && oi < redi[tid])) {
          red[tid] = ov; redi[tid] = oi;
        }
      }
      __syncthreads();
    }
    if (tid == 0) {
      int idx = redi[0];
      coords[(blockIdx.x * 16 + k) * 2 + 0] = idx % Wimg;  // x
      coords[(blockIdx.x * 16 + k) * 2 + 1] = idx / Wimg;  // y
      vals[idx] = -1e30f;
    }
    __syncthreads();
  }
}

__device__ __forceinline__ void roi_axis(int i, int start, int size,
                                         int& lo, int& hi, float& w)
{
  float sz = (float)size;
  float src = ((float)i + 0.5f) * sz * (1.f / 7.f) - 0.5f;
  src = fminf(fmaxf(src, 0.f), sz - 1.f);
  float fl = floorf(src);
  int l = (int)fl;
  lo = start + l;
  hi = start + min(l + 1, size - 1);
  w = src - fl;
}

// bilinear ROI extract -> roiR/roiT f32 and bf16 concat buffer (BA,7,7,512)
__global__ void extractk(const float* __restrict__ rgb, const float* __restrict__ tir,
                         const int* __restrict__ cR, const int* __restrict__ cT,
                         const float* __restrict__ sums,  // [0..3]=rgb, [4..7]=tir
                         float* __restrict__ roiR, float* __restrict__ roiT,
                         bf16* __restrict__ cat)
{
  const int ba = blockIdx.x;
  const int b = ba >> 4;
  const int pos = blockIdx.y;
  const int i = pos / 7, j = pos % 7;
  const bool useR = sums[b] > sums[4 + b];
  const int* cc = useR ? cR : cT;
  const int x = cc[ba * 2 + 0], y = cc[ba * 2 + 1];
  const int x0 = max(x - 3, 0), x1 = min(x + 4, 64);
  const int y0 = max(y - 3, 0), y1 = min(y + 4, 64);
  int yl, yh; float wy; roi_axis(i, y0, y1 - y0, yl, yh, wy);
  int xl, xh; float wx; roi_axis(j, x0, x1 - x0, xl, xh, wx);
  const int c = threadIdx.x;
  const size_t base = (size_t)b * 64 * 64 * 256;
  const size_t i00 = base + ((size_t)yl * 64 + xl) * 256 + c;
  const size_t i01 = base + ((size_t)yl * 64 + xh) * 256 + c;
  const size_t i10 = base + ((size_t)yh * 64 + xl) * 256 + c;
  const size_t i11 = base + ((size_t)yh * 64 + xh) * 256 + c;
  const float w00 = (1 - wy) * (1 - wx), w01 = (1 - wy) * wx;
  const float w10 = wy * (1 - wx), w11 = wy * wx;
  const float vR = w00 * rgb[i00] + w01 * rgb[i01] + w10 * rgb[i10] + w11 * rgb[i11];
  const float vT = w00 * tir[i00] + w01 * tir[i01] + w10 * tir[i10] + w11 * tir[i11];
  const size_t ro = ((size_t)ba * 49 + pos) * 256 + c;
  roiR[ro] = vR; roiT[ro] = vT;
  const size_t co = ((size_t)ba * 49 + pos) * 512;
  cat[co + c] = f2bf(vR);
  cat[co + 256 + c] = f2bf(vT);
}

// wg2 (64->2) + softmax + blend: fused = w0*roiR + w1*roiT
__global__ void combinek(const float* __restrict__ wg,  // (BA,49,64)
                         const float* __restrict__ w2,  // (64,2)
                         const float* __restrict__ b2,  // (2)
                         const float* __restrict__ roiR, const float* __restrict__ roiT,
                         float* __restrict__ fused)
{
  __shared__ float sw[2];
  const int ba = blockIdx.x, pos = blockIdx.y, c = threadIdx.x;
  if (c == 0) {
    const float* wp = wg + ((size_t)ba * 49 + pos) * 64;
    float l0 = b2[0], l1 = b2[1];
    for (int k = 0; k < 64; ++k) { float v = wp[k]; l0 += v * w2[k * 2]; l1 += v * w2[k * 2 + 1]; }
    float m = fmaxf(l0, l1);
    float e0 = __expf(l0 - m), e1 = __expf(l1 - m);
    float inv = 1.f / (e0 + e1);
    sw[0] = e0 * inv; sw[1] = e1 * inv;
  }
  __syncthreads();
  const size_t o = ((size_t)ba * 49 + pos) * 256 + c;
  fused[o] = sw[0] * roiR[o] + sw[1] * roiT[o];
}

// integrate: resize fused ROI back into its clipped rect. Reference processes
// anchors sequentially (later overwrite); equivalently, the highest-index
// anchor covering a pixel wins -> per-anchor blocks skip covered pixels.
__global__ void integratek(const float* __restrict__ fused,
                           const int* __restrict__ cR, const int* __restrict__ cT,
                           const float* __restrict__ sums,
                           float* __restrict__ rgb, float* __restrict__ tir)
{
  const int ba = blockIdx.x;
  const int b = ba >> 4, a = ba & 15;
  const bool useR = sums[b] > sums[4 + b];
  const int* cc = useR ? cR : cT;
  const int x = cc[ba * 2], y = cc[ba * 2 + 1];
  const int x0 = max(x - 3, 0), rw = min(x + 4, 64) - x0;
  const int y0 = max(y - 3, 0), rh = min(y + 4, 64) - y0;
  const int total = rh * rw * 256;
  for (int t = threadIdx.x; t < total; t += blockDim.x) {
    const int c = t & 255;
    const int p = t >> 8;
    const int v = p / rw, u = p % rw;
    const int yy = y0 + v, xx = x0 + u;
    bool covered = false;
    for (int a2 = a + 1; a2 < 16; ++a2) {
      const int x2 = cc[(b * 16 + a2) * 2], y2 = cc[(b * 16 + a2) * 2 + 1];
      const int xs = max(x2 - 3, 0), xe = min(x2 + 4, 64);
      const int ys = max(y2 - 3, 0), ye = min(y2 + 4, 64);
      if (yy >= ys && yy < ye && xx >= xs && xx < xe) { covered = true; break; }
    }
    if (covered) continue;
    float srcY = fminf(fmaxf(((float)v + 0.5f) * 7.f / (float)rh - 0.5f, 0.f), 6.f);
    float flY = floorf(srcY); int yl = (int)flY; int yh = min(yl + 1, 6); float wy = srcY - flY;
    float srcX = fminf(fmaxf(((float)u + 0.5f) * 7.f / (float)rw - 0.5f, 0.f), 6.f);
    float flX = floorf(srcX); int xl = (int)flX; int xh = min(xl + 1, 6); float wx = srcX - flX;
    const size_t fb = (size_t)ba * 49 * 256;
    const float v00 = fused[fb + ((size_t)yl * 7 + xl) * 256 + c];
    const float v01 = fused[fb + ((size_t)yl * 7 + xh) * 256 + c];
    const float v10 = fused[fb + ((size_t)yh * 7 + xl) * 256 + c];
    const float v11 = fused[fb + ((size_t)yh * 7 + xh) * 256 + c];
    const float val = (1 - wy) * (1 - wx) * v00 + (1 - wy) * wx * v01
                    + wy * (1 - wx) * v10 + wy * wx * v11;
    const size_t o = ((size_t)b * 64 * 64 + (size_t)yy * 64 + xx) * 256 + c;
    rgb[o] = val;
    tir[o] = val;
  }
}

// ---------------------------------------------------------------------------
// Host orchestration
// ---------------------------------------------------------------------------
enum {
  I_FRGB = 0, I_FTIR,
  AR_C1W, AR_C1B, AR_G1, AR_B1, AR_C2W, AR_C2B,
  AT_C1W, AT_C1B, AT_G1, AT_B1, AT_C2W, AT_C2B,
  LF_C1W, LF_C1B, LF_G1, LF_B1, LF_C2W, LF_C2B, LF_G2, LF_B2,
  LF_WG1W, LF_WG1B, LF_WG2W, LF_WG2B,
  FR_C1W, FR_C1B, FR_G1, FR_B1, FR_C2W, FR_C2B, FR_G2, FR_B2
};

static void conv(hipStream_t s, const bf16* x, int nImg, int Hh, int Ww, int Cin,
                 const bf16* wT, const float* bias, const float* g, const float* bb,
                 int Cout, int KH, int KW, int pad, int relu,
                 float* oF, bf16* oH, const float* res)
{
  const int HW = Hh * Ww;
  const int tiles = (HW + 63) / 64;
  dim3 grid(nImg * tiles, Cout / 64);
  conv_wmma<<<grid, 256, 0, s>>>(x, nImg, Hh, Ww, Cin, wT, bias, g, bb,
                                 Cout, KH, KW, pad, relu, oF, oH, res);
}

extern "C" void kernel_launch(void* const* d_in, const int* in_sizes, int n_in,
                              void* d_out, int out_size, void* d_ws, size_t ws_size,
                              hipStream_t stream)
{
  (void)in_sizes; (void)n_in; (void)out_size; (void)ws_size;
  const int B = 4, H = 64, W = 64, C = 256, C4 = 64, A = 16, R = 7;
  const size_t NPIX = (size_t)B * H * W;       // 16384
  const size_t FEAT = NPIX * C;                // 4,194,304
  const size_t NROI = (size_t)B * A * R * R;   // 3136

  char* ws = (char*)d_ws;
  size_t off = 0;
  auto alloc = [&](size_t bytes) -> void* {
    void* p = ws + off;
    off = (off + bytes + 255) & ~(size_t)255;
    return p;
  };

  float* curR = (float*)alloc(FEAT * 4);
  float* curT = (float*)alloc(FEAT * 4);
  float* curG = (float*)alloc(FEAT * 4);
  bf16*  xbf  = (bf16*) alloc(FEAT * 2);
  float* h1   = (float*)alloc(NPIX * C4 * 4);
  float* amap = (float*)alloc(NPIX * 4);
  float* sums = (float*)alloc(8 * 4);
  int* coordsR = (int*)alloc((size_t)B * A * 2 * 4);
  int* coordsT = (int*)alloc((size_t)B * A * 2 * 4);
  bf16* w_adR = (bf16*)alloc((size_t)9 * C * C4 * 2);
  bf16* w_adT = (bf16*)alloc((size_t)9 * C * C4 * 2);
  bf16* w_lf1 = (bf16*)alloc((size_t)9 * 2 * C * C * 2);
  bf16* w_lf2 = (bf16*)alloc((size_t)9 * C * C * 2);
  bf16* w_wg1 = (bf16*)alloc((size_t)C * C4 * 2);
  bf16* w_fr1 = (bf16*)alloc((size_t)9 * 2 * C * C * 2);
  bf16* w_fr2 = (bf16*)alloc((size_t)C * C * 2);
  bf16*  catlf = (bf16*) alloc(NROI * 2 * C * 2);
  float* roiR  = (float*)alloc(NROI * C * 4);
  float* roiT  = (float*)alloc(NROI * C * 4);
  bf16*  lfhA  = (bf16*) alloc(NROI * C * 2);
  bf16*  lfhB  = (bf16*) alloc(NROI * C * 2);
  float* wgb   = (float*)alloc(NROI * C4 * 4);
  float* fus   = (float*)alloc(NROI * C * 4);
  bf16*  catfr = (bf16*) alloc(FEAT * 2 * 2);
  bf16*  frh   = (bf16*) alloc(FEAT * 2);

  auto F = [&](int i) { return (const float*)d_in[i]; };

  (void)hipMemcpyAsync(curR, d_in[I_FRGB], FEAT * 4, hipMemcpyDeviceToDevice, stream);
  (void)hipMemcpyAsync(curT, d_in[I_FTIR], FEAT * 4, hipMemcpyDeviceToDevice, stream);

  auto WP = [&](int idx, bf16* dst, int taps, int Cin, int Cout) {
    long tot = (long)taps * Cin * Cout;
    wprep<<<(unsigned)((tot + 255) / 256), 256, 0, stream>>>(F(idx), dst, Cin, Cout, tot);
  };
  WP(AR_C1W, w_adR, 9, C, C4);
  WP(AT_C1W, w_adT, 9, C, C4);
  WP(LF_C1W, w_lf1, 9, 2 * C, C);
  WP(LF_C2W, w_lf2, 9, C, C);
  WP(LF_WG1W, w_wg1, 1, C, C4);
  WP(FR_C1W, w_fr1, 9, 2 * C, C);
  WP(FR_C2W, w_fr2, 1, C, C);

  const unsigned gFeat = (unsigned)((FEAT + 255) / 256);
  dim3 eg(B * A, R * R);

  for (int it = 0; it < 2; ++it) {
    // --- RGB detector ---
    tobf<<<gFeat, 256, 0, stream>>>(curR, xbf, (long)FEAT);
    conv(stream, xbf, B, H, W, C, w_adR, F(AR_C1B), F(AR_G1), F(AR_B1),
         C4, 3, 3, 1, 1, h1, nullptr, nullptr);
    detheadk<<<(unsigned)(NPIX / 256), 256, 0, stream>>>(h1, F(AR_C2W), F(AR_C2B), amap);
    topkk<<<B, 256, 0, stream>>>(amap, H * W, W, coordsR, sums + 0);
    // --- TIR detector ---
    tobf<<<gFeat, 256, 0, stream>>>(curT, xbf, (long)FEAT);
    conv(stream, xbf, B, H, W, C, w_adT, F(AT_C1B), F(AT_G1), F(AT_B1),
         C4, 3, 3, 1, 1, h1, nullptr, nullptr);
    detheadk<<<(unsigned)(NPIX / 256), 256, 0, stream>>>(h1, F(AT_C2W), F(AT_C2B), amap);
    topkk<<<B, 256, 0, stream>>>(amap, H * W, W, coordsT, sums + 4);
    // --- ROI extract + concat ---
    extractk<<<eg, 256, 0, stream>>>(curR, curT, coordsR, coordsT, sums,
                                     roiR, roiT, catlf);
    // --- local fusion ---
    conv(stream, catlf, B * A, R, R, 2 * C, w_lf1, F(LF_C1B), F(LF_G1), F(LF_B1),
         C, 3, 3, 1, 1, nullptr, lfhA, nullptr);
    conv(stream, lfhA, B * A, R, R, C, w_lf2, F(LF_C2B), F(LF_G2), F(LF_B2),
         C, 3, 3, 1, 1, nullptr, lfhB, nullptr);
    conv(stream, lfhB, B * A, R, R, C, w_wg1, F(LF_WG1B), nullptr, nullptr,
         C4, 1, 1, 0, 1, wgb, nullptr, nullptr);
    combinek<<<eg, 256, 0, stream>>>(wgb, F(LF_WG2W), F(LF_WG2B), roiR, roiT, fus);
    // --- integrate (in place on curR/curT) ---
    integratek<<<B * A, 256, 0, stream>>>(fus, coordsR, coordsT, sums, curR, curT);
    if (it == 0) {
      avgk<<<gFeat, 256, 0, stream>>>(curR, curT, curG, (long)FEAT);
    }
  }

  // --- final fusion-refine ---
  catw<<<gFeat, 256, 0, stream>>>(curG, catfr, C, 2 * C, 0, (long)FEAT);
  catw<<<gFeat, 256, 0, stream>>>(curR, catfr, C, 2 * C, C, (long)FEAT);
  conv(stream, catfr, B, H, W, 2 * C, w_fr1, F(FR_C1B), F(FR_G1), F(FR_B1),
       C, 3, 3, 1, 1, nullptr, frh, nullptr);
  conv(stream, frh, B, H, W, C, w_fr2, F(FR_C2B), F(FR_G2), F(FR_B2),
       C, 1, 1, 0, 0, (float*)d_out, nullptr, curG);
}
